// TemplMatch_27736898798201
// MI455X (gfx1250) — matrix operand order repaired and checked by reading, not compiled, not run on hardware
//
#include <hip/hip_runtime.h>

typedef _Float16 v4h  __attribute__((ext_vector_type(4)));
typedef _Float16 v8h  __attribute__((ext_vector_type(8)));
typedef _Float16 v16h __attribute__((ext_vector_type(16)));
typedef float    v8f  __attribute__((ext_vector_type(8)));
typedef int      v4i  __attribute__((ext_vector_type(4)));

#define NIMG   128
#define HW     512
#define KDIM   472
#define OW     41            // 512-472+1
#define PADR   528           // padded rows of f16 image (need 519)
#define PADC   544           // padded cols of f16 image (need 544 exactly)
#define XFIMG  (PADR*PADC)   // 287232 halves per image
#define TILES  (NIMG*3*3)    // 1152 C-tiles (n, x0-tile, y0-tile)
#define KSEG   8             // K-split for occupancy
#define KHPSEG 59            // 472/8
#define BKHH   8192          // prepB halves per kh (16 chunks * 32 lanes * 16 halves)
#define OUTSZ  (NIMG*OW*OW)  // 215168

#if defined(__has_builtin)
#if __has_builtin(__builtin_amdgcn_global_load_async_to_lds_b128) && \
    __has_builtin(__builtin_amdgcn_s_wait_asynccnt)
#define USE_ASYNC_LDS 1
#endif
#endif

#ifdef USE_ASYNC_LDS
typedef __attribute__((address_space(1))) v4i* gv4i_p;
typedef __attribute__((address_space(3))) v4i* lv4i_p;
__device__ __forceinline__ void async_cp16(const void* g, void* l) {
    __builtin_amdgcn_global_load_async_to_lds_b128((gv4i_p)g, (lv4i_p)l, 0, 0);
}
#endif

// ---------------- prep: X fp32 -> zero-padded fp16 image buffer ----------------
__global__ void k_prep_x(const float* __restrict__ X, _Float16* __restrict__ xf) {
    long idx = (long)blockIdx.x * 256 + threadIdx.x;           // over NIMG*PADR*(PADC/4)
    int c4 = (int)(idx % (PADC / 4));
    long t  = idx / (PADC / 4);
    int r  = (int)(t % PADR);
    int n  = (int)(t / PADR);
    int c0 = c4 * 4;
    v4h v;
#pragma unroll
    for (int i = 0; i < 4; ++i) {
        int c = c0 + i;
        v[i] = (r < HW && c < HW) ? (_Float16)X[((long)n * HW + r) * HW + c] : (_Float16)0.0f;
    }
    *(v4h*)(xf + (long)n * XFIMG + (long)r * PADC + c0) = v;
}

// ------- prep: template -> lane-exact prepacked B fragments (shifted rows) -----
// Layout per kh: [chunk c(16)][lane l(32)][h(16)] halves; value = T[kh][w],
// w = c*32 + ((l>=16)?16:0) + h - (l&15), zero outside [0,472).
__global__ void k_prep_b(const float* __restrict__ Kern, _Float16* __restrict__ prepB) {
    long idx = (long)blockIdx.x * 256 + threadIdx.x;           // over 472*8192
    int kh  = (int)(idx >> 13);
    int rem = (int)(idx & 8191);
    int c = rem >> 9;
    int l = (rem >> 4) & 31;
    int h = rem & 15;
    int w = c * 32 + ((l >> 4) << 4) + h - (l & 15);
    _Float16 v = (_Float16)0.0f;
    if (w >= 0 && w < KDIM) v = (_Float16)Kern[(long)kh * KDIM + w];
    prepB[idx] = v;
}

// --------------- box-filter stage 1: per-row sliding-window sums ---------------
__global__ void k_rowsum(const float* __restrict__ X,
                         float* __restrict__ rs, float* __restrict__ rs2) {
    int t = blockIdx.x * 256 + threadIdx.x;                    // n*512 + y
    const float* row = X + (long)t * HW;
    float s = 0.f, s2 = 0.f;
    for (int w = 0; w < KDIM; ++w) { float v = row[w]; s += v; s2 += v * v; }
    rs[(long)t * OW]  = s;
    rs2[(long)t * OW] = s2;
    for (int x = 1; x < OW; ++x) {
        float a = row[x + KDIM - 1], b = row[x - 1];
        s  += a - b;
        s2 += a * a - b * b;
        rs[(long)t * OW + x]  = s;
        rs2[(long)t * OW + x] = s2;
    }
}

// ----------------------------- main WMMA correlation ---------------------------
__global__ void __launch_bounds__(256)
k_corr(const _Float16* __restrict__ xf, const _Float16* __restrict__ prepB,
       float* __restrict__ part) {
    __shared__ _Float16 ldsB[2 * BKHH];                        // 32 KB double buffer

    const int kseg = blockIdx.x & (KSEG - 1);
    const int tg   = blockIdx.x / KSEG;
    const int wave = threadIdx.x >> 5;
    const int lane = threadIdx.x & 31;
    const int tile = tg * 8 + wave;                            // [0,1152)
    const int n  = tile / 9;
    const int rr = tile % 9;
    const int x0 = (rr / 3) * 16;
    const int y0 = (rr % 3) * 16;
    const int kh0 = kseg * KHPSEG;

    const _Float16* Abase = xf + (long)n * XFIMG
                          + (long)(y0 + (lane & 15)) * PADC
                          + x0 + ((lane & 16) ? 16 : 0);
    const _Float16* Bseg = prepB + (long)kh0 * BKHH;
    const int t = threadIdx.x;

    // stage B for first kh of this segment into buffer 0
    {
        const uint4* g = (const uint4*)Bseg;                   // 1024 uint4 = 16KB
        uint4* l = (uint4*)ldsB;
#pragma unroll
        for (int i = 0; i < 4; ++i) {
#ifdef USE_ASYNC_LDS
            async_cp16(g + t * 4 + i, l + t * 4 + i);
#else
            l[t * 4 + i] = g[t * 4 + i];
#endif
        }
    }
#ifdef USE_ASYNC_LDS
    __builtin_amdgcn_s_wait_asynccnt(0);
#endif
    __syncthreads();

    v8f acc = {0.f, 0.f, 0.f, 0.f, 0.f, 0.f, 0.f, 0.f};

    for (int kk = 0; kk < KHPSEG; ++kk) {
        // prefetch next kh's B into the other buffer
        if (kk + 1 < KHPSEG) {
            const uint4* g = (const uint4*)(Bseg + (long)(kk + 1) * BKHH);
            uint4* l = (uint4*)(ldsB + ((kk + 1) & 1) * BKHH);
#pragma unroll
            for (int i = 0; i < 4; ++i) {
#ifdef USE_ASYNC_LDS
                async_cp16(g + t * 4 + i, l + t * 4 + i);
#else
                l[t * 4 + i] = g[t * 4 + i];
#endif
            }
        }

        const _Float16* Ak = Abase + (long)(kh0 + kk) * PADC;
        const _Float16* Bl = ldsB + (kk & 1) * BKHH + (lane << 4);
#pragma unroll
        for (int c = 0; c < 16; ++c) {
            v8h alo = *(const v8h*)(Ak + c * 32);
            v8h ahi = *(const v8h*)(Ak + c * 32 + 8);
            v8h blo = *(const v8h*)(Bl + c * 512);
            v8h bhi = *(const v8h*)(Bl + c * 512 + 8);
            v16h a = __builtin_shufflevector(alo, ahi, 0,1,2,3,4,5,6,7,8,9,10,11,12,13,14,15);
            v16h b = __builtin_shufflevector(blo, bhi, 0,1,2,3,4,5,6,7,8,9,10,11,12,13,14,15);
            acc = __builtin_amdgcn_wmma_f32_16x16x32_f16(
                false, a, false, b, (short)0, acc, false, false);
        }

#ifdef USE_ASYNC_LDS
        __builtin_amdgcn_s_wait_asynccnt(0);
#endif
        __syncthreads();
    }

    // store partial numerator: VGPR r -> M = r + (lane>=16?8:0), N = lane&15
    float* P = part + (long)kseg * OUTSZ + (long)n * OW * OW;
    const int x = x0 + (lane & 15);
    const int mofs = (lane & 16) ? 8 : 0;
#pragma unroll
    for (int r = 0; r < 8; ++r) {
        int y = y0 + r + mofs;
        if (y < OW && x < OW) P[(long)y * OW + x] = acc[r];
    }
}

// ------------- finalize: column box-sum, variance, divide, NaN->0 --------------
__global__ void k_finalize(const float* __restrict__ rs, const float* __restrict__ rs2,
                           const float* __restrict__ part,
                           const float* __restrict__ normp, const int* __restrict__ tnump,
                           float* __restrict__ out) {
    int idx = blockIdx.x * 256 + threadIdx.x;
    if (idx >= OUTSZ) return;
    int x  = idx % OW;
    int yo = (idx / OW) % OW;
    int n  = idx / (OW * OW);
    const float* rp  = rs  + ((long)n * HW + yo) * OW + x;
    const float* rp2 = rs2 + ((long)n * HW + yo) * OW + x;
    float ls = 0.f, ls2 = 0.f;
    for (int kh = 0; kh < KDIM; ++kh) {
        ls  += rp[(long)kh * OW];
        ls2 += rp2[(long)kh * OW];
    }
    float tn = (float)tnump[0];
    float varr = ls2 - ls * ls / tn + 1e-8f;
    varr = (varr < 0.f) ? 0.f : varr;
    float denom = sqrtf(normp[0] * varr);
    float num = 0.f;
#pragma unroll
    for (int s = 0; s < KSEG; ++s) num += part[(long)s * OUTSZ + idx];
    float o = num / denom;
    o = (o != o) ? 0.f : o;
    out[idx] = o;
}

// ----------------------------------- launch ------------------------------------
extern "C" void kernel_launch(void* const* d_in, const int* in_sizes, int n_in,
                              void* d_out, int out_size, void* d_ws, size_t ws_size,
                              hipStream_t stream) {
    const float* X    = (const float*)d_in[0];   // 128*512*512
    const float* Kern = (const float*)d_in[1];   // 472*472
    const float* norm = (const float*)d_in[2];   // 1
    const int*   tnum = (const int*)d_in[3];     // 1
    float* out = (float*)d_out;

    char* ws = (char*)d_ws;
    _Float16* xf    = (_Float16*)(ws);                         // 73,531,392 B
    _Float16* prepB = (_Float16*)(ws + 73531392);              //  7,733,248 B
    float*    rs    = (float*)   (ws + 81264640);              // 10,747,904 B
    float*    rs2   = (float*)   (ws + 92012544);              // 10,747,904 B
    float*    part  = (float*)   (ws + 102760448);             //  6,885,376 B
    (void)in_sizes; (void)n_in; (void)out_size; (void)ws_size;

    k_prep_x <<<dim3((NIMG * PADR * (PADC / 4)) / 256), dim3(256), 0, stream>>>(X, xf);
    k_prep_b <<<dim3((KDIM * BKHH) / 256),              dim3(256), 0, stream>>>(Kern, prepB);
    k_rowsum <<<dim3((NIMG * HW) / 256),                dim3(256), 0, stream>>>(X, rs, rs2);
    k_corr   <<<dim3((TILES / 8) * KSEG),               dim3(256), 0, stream>>>(xf, prepB, part);
    k_finalize<<<dim3((OUTSZ + 255) / 256),             dim3(256), 0, stream>>>(rs, rs2, part,
                                                                               norm, tnum, out);
}
